// WithinImageLoss1_29746943492121
// MI455X (gfx1250) — matrix operand order repaired and checked by reading, not compiled, not run on hardware
//
#include <hip/hip_runtime.h>

// ---- problem constants (from reference) ----
#define C_DIM   128
#define P_DIM   4225          // 65*65
#define INV_T   (1.0f / 0.07f)
#define N1_F    (1.0f / 4225.0f)

// ---- tiling ----
#define KEYT    64            // keys staged per LDS tile
#define QPW     16            // queries per wave
#define WAVES   2             // waves per workgroup
#define BLOCK   (WAVES * 32)  // 64 threads
#define QPB     (WAVES * QPW) // 32 queries per WG
#define NQB     ((P_DIM + QPB - 1) / QPB)   // 133 workgroups
#define NKT     ((P_DIM + KEYT - 1) / KEYT) // 67 key tiles
#define STR     136           // LDS row stride in bf16 elems (256B data + 16B pad = 272B)

typedef __attribute__((ext_vector_type(16))) __bf16        v16bf;
typedef __attribute__((ext_vector_type(8)))  float         v8f;
typedef __attribute__((ext_vector_type(4)))  unsigned int  v4u;
typedef __attribute__((ext_vector_type(8)))  int           v8i;
typedef __attribute__((ext_vector_type(4)))  int           v4i;

struct B32x2 { uint4 lo, hi; };

// round-to-nearest-even f32 -> bf16 bits
static __device__ __forceinline__ unsigned short f2bf(float f) {
    unsigned int u = __float_as_uint(f);
    u += 0x7FFFu + ((u >> 16) & 1u);
    return (unsigned short)(u >> 16);
}

// assemble a 32-byte WMMA fragment from two 16-byte chunks (LDS or global)
static __device__ __forceinline__ v16bf ld_frag(const unsigned short* p0,
                                                const unsigned short* p1) {
    B32x2 t;
    t.lo = *(const uint4*)p0;
    t.hi = *(const uint4*)p1;
    return __builtin_bit_cast(v16bf, t);
}

// ---- TDM: DMA one [rows x 128ch] bf16 tile (row-contiguous in global) into LDS,
//      with hardware row padding 256B data + 16B pad -> LDS row stride 272B.
//      tensor_dim1 = remaining rows => TDM zero-fills past-the-end rows.
static __device__ __forceinline__ void tdm_load_tile(unsigned lds_off,
                                                     const unsigned short* gptr,
                                                     unsigned rows_rem) {
    unsigned long long ga = (unsigned long long)(uintptr_t)gptr;
    v4u g0;
    g0[0] = 1u;                                  // count=1, user mode, no gather
    g0[1] = lds_off;                             // LDS byte address
    g0[2] = (unsigned)ga;                        // global_addr[31:0]
    g0[3] = ((unsigned)(ga >> 32) & 0x01FFFFFFu) // global_addr[56:32]
            | 0x80000000u;                       // type=2 ("image")
    v8i g1;
    g1[0] = (int)((1u << 16)     // data_size = 2 bytes
                | (1u << 20)     // pad_enable
                | (5u << 22)     // pad_interval: 8*2^5 = 256 bytes
                | (3u << 25));   // pad_amount: 4 DWORDs = 16 bytes
    g1[1] = (int)(128u << 16);                         // tensor_dim0 = 128
    g1[2] = (int)((rows_rem & 0xFFFFu) << 16);         // tensor_dim1 lo16
    g1[3] = (int)((rows_rem >> 16) | (128u << 16));    // tensor_dim1 hi16 | tile_dim0=128
    g1[4] = KEYT;                                      // tile_dim1 = 64, tile_dim2 = 0
    g1[5] = 128;                                       // tensor_dim0_stride = 128
    g1[6] = 0;
    g1[7] = 0;
    v4i z = {0, 0, 0, 0};                              // 2D tensor: groups 2/3 unused
#if defined(__clang_major__) && (__clang_major__ >= 23)
    v8i z8 = {0, 0, 0, 0, 0, 0, 0, 0};
    __builtin_amdgcn_tensor_load_to_lds(g0, g1, z, z, z8, 0);
#else
    __builtin_amdgcn_tensor_load_to_lds(g0, g1, z, z, 0);
#endif
}

// -------- kernel 0: count positives, zero the output accumulator --------
__global__ __launch_bounds__(256) void nce_init_kernel(const int* __restrict__ x1o,
                                                       float* __restrict__ ws,
                                                       float* __restrict__ out) {
    __shared__ int red[256];
    int t = threadIdx.x;
    int s = 0;
    for (int i = t; i < P_DIM; i += 256) s += (x1o[i] == 1) ? 1 : 0;
    red[t] = s;
    __syncthreads();
    for (int off = 128; off > 0; off >>= 1) {
        if (t < off) red[t] += red[t + off];
        __syncthreads();
    }
    if (t == 0) { ws[0] = (float)red[0]; out[0] = 0.0f; }
}

// -------- kernel 1: one-shot f32 [c][p] -> bf16 [p][c] transpose-convert --------
// Queries are pre-scaled by 1/TEMP so the WMMA accumulator is tt directly.
__global__ __launch_bounds__(256) void nce_convert_kernel(const float* __restrict__ x0,
                                                          const float* __restrict__ x1,
                                                          unsigned short* __restrict__ xq,
                                                          unsigned short* __restrict__ xk) {
    const int n = C_DIM * P_DIM;
    for (int idx = blockIdx.x * blockDim.x + threadIdx.x; idx < n;
         idx += gridDim.x * blockDim.x) {
        int c = idx / P_DIM;
        int p = idx - c * P_DIM;
        xq[p * C_DIM + c] = f2bf(x0[idx] * INV_T);
        xk[p * C_DIM + c] = f2bf(x1[idx]);
    }
}

// -------- kernel 2 (TDM path): fused WMMA gram + online logsumexp --------
__global__ __launch_bounds__(BLOCK) void nce_main_tdm(const unsigned short* __restrict__ xq,
                                                      const unsigned short* __restrict__ xk,
                                                      const int*   __restrict__ x0o,
                                                      const int*   __restrict__ x1o,
                                                      const float* __restrict__ ws,
                                                      float*       __restrict__ out) {
    __shared__ unsigned short ldsK[2][KEYT * STR];          // double-buffered key tiles
    __shared__ __align__(8) unsigned char ldsM[2][KEYT];    // 0=OOB, 1=key, 2=positive

    const int tid  = threadIdx.x;
    const int lane = tid & 31;
    const int w    = tid >> 5;
    const int h    = lane >> 4;
    const int ln16 = lane & 15;
    const int qb   = blockIdx.x * QPB;

    // ---- loop-invariant query B-fragments straight from global (L2-resident) ----
    // B 32x16: lanes 0-15 hold K=0..15, lanes 16-31 hold K=16..31; N = lane%16
    int qrow = qb + w * QPW + ln16;
    if (qrow >= P_DIM) qrow = P_DIM - 1;
    const unsigned short* qp = xq + (size_t)qrow * C_DIM;
    v16bf bq[4];
    #pragma unroll
    for (int cb = 0; cb < 4; ++cb) {
        const unsigned short* p = qp + cb * 32 + (h << 4);
        bq[cb] = ld_frag(p, p + 8);
    }

    // ---- prologue: stage mask tile 0; wave0 kicks off TDM for tile 0 ----
    ldsM[0][tid] = (tid < P_DIM) ? ((x1o[tid] == 1) ? 2 : 1) : 0;
    if (tid < 32)
        tdm_load_tile((unsigned)(uintptr_t)&ldsK[0][0], xk, (unsigned)P_DIM);

    float m0  = -__builtin_inff();
    float s0  = 0.0f;
    float ms0 = 0.0f;

    for (int t = 0; t < NKT; ++t) {
        const int cur = t & 1;
        if (tid < 32) __builtin_amdgcn_s_wait_tensorcnt(0);  // wave0: tile t arrived
        __syncthreads();  // tile t visible to all; buf[cur^1] free to overwrite

        // ---- prefetch tile t+1 into the other buffer (overlaps compute) ----
        if (t + 1 < NKT) {
            const int kb2 = (t + 1) * KEYT;
            const int gk  = kb2 + tid;
            ldsM[cur ^ 1][tid] = (gk < P_DIM) ? ((x1o[gk] == 1) ? 2 : 1) : 0;
            if (tid < 32)
                tdm_load_tile((unsigned)(uintptr_t)&ldsK[cur ^ 1][0],
                              xk + (size_t)kb2 * C_DIM, (unsigned)(P_DIM - kb2));
        }

        // ---- compute: 4 sub-tiles of 16 keys ----
        #pragma unroll
        for (int ko = 0; ko < KEYT; ko += 16) {
            const unsigned long long mb =
                *(const unsigned long long*)&ldsM[cur][ko + (h << 3)];

            v8f acc = {};
            #pragma unroll
            for (int cb = 0; cb < 4; ++cb) {
                // A 16x32: lane row M=lane%16; elems 0..15 -> K=16h+0..15
                const unsigned short* p =
                    &ldsK[cur][(ko + ln16) * STR + cb * 32 + (h << 4)];
                v16bf a = ld_frag(p, p + 8);
                acc = __builtin_amdgcn_wmma_f32_16x16x32_bf16(
                          false, a, false, bq[cb], (short)0, acc, false, false);
            }

            // ---- branchless online softmax + masked-sum (acc is tt already) ----
            float tt[8];
            float mnew = m0;
            #pragma unroll
            for (int r = 0; r < 8; ++r) {
                unsigned int mk = (unsigned int)((mb >> (8 * r)) & 0xFFu);
                float v = acc[r];
                ms0 += (mk == 2u) ? v : 0.0f;
                tt[r] = mk ? v : -__builtin_inff();
                mnew = fmaxf(mnew, tt[r]);
            }
            // m0 finite after sub-tile 0 (keys 0..63 all valid); exp(-inf)=0 covers the rest
            s0 *= __expf(m0 - mnew);
            #pragma unroll
            for (int r = 0; r < 8; ++r) s0 += __expf(tt[r] - mnew);
            m0 = mnew;
        }
    }

    // ---- merge half-wave partials (lane <-> lane^16), finish loss ----
    float m1  = __shfl_xor(m0,  16, 32);
    float s1  = __shfl_xor(s0,  16, 32);
    float ms1 = __shfl_xor(ms0, 16, 32);
    float M = fmaxf(m0, m1);
    float S = s0 * __expf(m0 - M) + s1 * __expf(m1 - M);
    float lse  = M + __logf(S);
    float msum = ms0 + ms1;

    const float npos = ws[0];
    const int q = qb + w * QPW + ln16;
    float sum2 = msum - npos * lse;
    bool valid = (q < P_DIM) && (x0o[q] != 0);
    float contrib = valid ? sum2 : 0.0f;

    #pragma unroll
    for (int off = 1; off < 16; off <<= 1)
        contrib += __shfl_xor(contrib, off, 32);

    if (lane == 0)
        atomicAdd(out, -N1_F * npos * contrib);
}

// -------- fallback (small workspace): on-the-fly conversion staging --------
__global__ __launch_bounds__(BLOCK) void nce_main_fallback(const float* __restrict__ x0,
                                                           const float* __restrict__ x1,
                                                           const int*   __restrict__ x0o,
                                                           const int*   __restrict__ x1o,
                                                           const float* __restrict__ ws,
                                                           float*       __restrict__ out) {
    __shared__ unsigned short ldsK[KEYT * STR];
    __shared__ unsigned short ldsQ[QPB  * STR];
    __shared__ __align__(8) unsigned char ldsM[KEYT + 16];

    const int tid  = threadIdx.x;
    const int lane = tid & 31;
    const int w    = tid >> 5;
    const int h    = lane >> 4;
    const int ln16 = lane & 15;
    const int qb   = blockIdx.x * QPB;

    for (int e = tid; e < C_DIM * QPB; e += BLOCK) {
        int c = e >> 5;
        int q = e & (QPB - 1);
        int gq = qb + q; if (gq >= P_DIM) gq = P_DIM - 1;
        ldsQ[q * STR + c] = f2bf(x0[c * P_DIM + gq] * INV_T);  // pre-scaled by 1/TEMP
    }
    __syncthreads();

    v16bf bq[4];
    {
        const int qrow = w * QPW + ln16;
        #pragma unroll
        for (int cb = 0; cb < 4; ++cb) {
            const unsigned short* p = &ldsQ[qrow * STR + cb * 32 + (h << 4)];
            bq[cb] = ld_frag(p, p + 8);
        }
    }

    float m0 = -__builtin_inff();
    float s0 = 0.0f;
    float ms0 = 0.0f;

    for (int tk = 0; tk < NKT; ++tk) {
        const int kb = tk * KEYT;
        __syncthreads();

        for (int e = tid; e < C_DIM * KEYT; e += BLOCK) {
            int c = e >> 6;
            int k = e & (KEYT - 1);
            int gk = kb + k; if (gk >= P_DIM) gk = P_DIM - 1;
            ldsK[k * STR + c] = f2bf(x1[c * P_DIM + gk]);
        }
        {
            int gk = kb + tid;
            ldsM[tid] = (gk < P_DIM) ? ((x1o[gk] == 1) ? 2 : 1) : 0;
        }
        __syncthreads();

        #pragma unroll
        for (int ko = 0; ko < KEYT; ko += 16) {
            const unsigned long long mb =
                *(const unsigned long long*)&ldsM[ko + (h << 3)];

            v8f acc = {};
            #pragma unroll
            for (int cb = 0; cb < 4; ++cb) {
                const unsigned short* p =
                    &ldsK[(ko + ln16) * STR + cb * 32 + (h << 4)];
                v16bf a = ld_frag(p, p + 8);
                acc = __builtin_amdgcn_wmma_f32_16x16x32_bf16(
                          false, a, false, bq[cb], (short)0, acc, false, false);
            }

            float tt[8];
            float mnew = m0;
            #pragma unroll
            for (int r = 0; r < 8; ++r) {
                unsigned int mk = (unsigned int)((mb >> (8 * r)) & 0xFFu);
                float v = acc[r];
                ms0 += (mk == 2u) ? v : 0.0f;
                tt[r] = mk ? v : -__builtin_inff();
                mnew = fmaxf(mnew, tt[r]);
            }
            s0 *= __expf(m0 - mnew);
            #pragma unroll
            for (int r = 0; r < 8; ++r) s0 += __expf(tt[r] - mnew);
            m0 = mnew;
        }
    }

    float m1  = __shfl_xor(m0,  16, 32);
    float s1  = __shfl_xor(s0,  16, 32);
    float ms1 = __shfl_xor(ms0, 16, 32);
    float M = fmaxf(m0, m1);
    float S = s0 * __expf(m0 - M) + s1 * __expf(m1 - M);
    float lse  = M + __logf(S);
    float msum = ms0 + ms1;

    const float npos = ws[0];
    const int q = qb + w * QPW + ln16;
    float sum2 = msum - npos * lse;
    bool valid = (q < P_DIM) && (x0o[q] != 0);
    float contrib = valid ? sum2 : 0.0f;

    #pragma unroll
    for (int off = 1; off < 16; off <<= 1)
        contrib += __shfl_xor(contrib, off, 32);

    if (lane == 0)
        atomicAdd(out, -N1_F * npos * contrib);
}

extern "C" void kernel_launch(void* const* d_in, const int* in_sizes, int n_in,
                              void* d_out, int out_size, void* d_ws, size_t ws_size,
                              hipStream_t stream) {
    const float* x0  = (const float*)d_in[0];
    const float* x1  = (const float*)d_in[1];
    const int*   x0o = (const int*)d_in[2];
    const int*   x1o = (const int*)d_in[3];
    float* out = (float*)d_out;
    float* ws  = (float*)d_ws;

    nce_init_kernel<<<1, 256, 0, stream>>>(x1o, ws, out);

    const size_t SZT  = (size_t)P_DIM * C_DIM * sizeof(unsigned short); // 1,081,600 B
    const size_t NEED = 256 + 2 * SZT;

    if (ws_size >= NEED) {
        unsigned short* xq = (unsigned short*)((char*)d_ws + 256);
        unsigned short* xk = xq + (size_t)P_DIM * C_DIM;
        nce_convert_kernel<<<512, 256, 0, stream>>>(x0, x1, xq, xk);
        nce_main_tdm<<<NQB, BLOCK, 0, stream>>>(xq, xk, x0o, x1o, ws, out);
    } else {
        nce_main_fallback<<<NQB, BLOCK, 0, stream>>>(x0, x1, x0o, x1o, ws, out);
    }
}
